// SSM_cha_1305670058665
// MI455X (gfx1250) — matrix operand order repaired and checked by reading, not compiled, not run on hardware
//
#include <hip/hip_runtime.h>
#include <hip/hip_bf16.h>

// ---------------- problem constants ----------------
#define B_   16
#define D_   128           // D_MODEL == D_INNER
#define HW_  96
#define L_   (HW_*HW_)     // 9216 tokens per batch
#define NTOK (B_*L_)       // 147456
#define DTRANK 8
#define DSTATE 2
#define OUTC 128
#define NCHUNK 64
#define CL   (L_/NCHUNK)   // 144

typedef __bf16 bf16;
typedef bf16  v16bf __attribute__((ext_vector_type(16)));
typedef bf16  v8bf  __attribute__((ext_vector_type(8)));
typedef bf16  v4bf  __attribute__((ext_vector_type(4)));
typedef float v8f   __attribute__((ext_vector_type(8)));

__device__ __forceinline__ bf16 to_bf16(float f) {
  unsigned u = __builtin_bit_cast(unsigned, f);
  unsigned r = (u + 0x7FFFu + ((u >> 16) & 1u)) >> 16;
  return __builtin_bit_cast(bf16, (unsigned short)r);
}

// silu via raw v_rcp_f32 (no IEEE div refinement sequence)
__device__ __forceinline__ float fast_silu(float v) {
  return v * __builtin_amdgcn_rcpf(1.0f + __expf(-v));
}

// =====================================================================
// K0: pre-pack GEMM weights to bf16 (once per launch; tiny)
// =====================================================================
__global__ __launch_bounds__(256) void k_prep(const float* __restrict__ w_in,
                                              const float* __restrict__ w_out,
                                              bf16* __restrict__ wb_in,
                                              bf16* __restrict__ wb_out) {
  const int i = blockIdx.x * 256 + threadIdx.x;     // grid covers 256*128
  if (i < 256 * 128) wb_in[i] = to_bf16(w_in[i]);
  if (i < 128 * 128) wb_out[i] = to_bf16(w_out[i]);
}

// =====================================================================
// K1: xz = tokens @ in_proj_w^T  (WMMA bf16, fp32 acc), 32-token M tile
//   Stage 1: raw f32 A tile -> LDS via GLOBAL_LOAD_ASYNC_TO_LDS_B32
//   Stage 2: one conversion pass -> bf16 LDS in fragment order
//   WMMA loop: A frag = one contiguous 32B LDS read; B = 32B global read
//   half 0 -> x_in stored channel-major (B,D,L)  (feeds depthwise conv)
//   half 1 -> silu(z) stored token-major (B*L,D) (feeds gating)
// =====================================================================
__global__ __launch_bounds__(256) void k_in_proj(const float* __restrict__ x,
                                                 const bf16* __restrict__ wb,
                                                 float* __restrict__ xin_ch,
                                                 float* __restrict__ z_tok) {
  __shared__ float sAf[128 * 32];           // [k][m] raw tile: 32 tokens x 128 K
  __shared__ bf16  sAb[128 * 32];           // fragment-ordered bf16 tile
  const int tile = blockIdx.x;              // 32-token tile (L%32==0)
  const int half = blockIdx.y;              // 0: x_in half, 1: z half
  const int g0   = tile * 32;
  const int b    = g0 / L_;
  const int l0   = g0 % L_;
  const int tid  = threadIdx.x;

  // ---- stage 1: async copy global f32 -> LDS (16 x b32 per thread) ----
  {
    const float* xbase = x + ((size_t)b * D_) * L_ + l0;       // block-uniform base
    unsigned lds_base = (unsigned)(size_t)(&sAf[0]);           // LDS byte offset
    for (int idx = tid; idx < 4096; idx += 256) {
      int k = idx >> 5, m = idx & 31;
      unsigned goff = (unsigned)((k * L_ + m) * 4);
      unsigned loff = lds_base + (unsigned)idx * 4u;
      asm volatile("global_load_async_to_lds_b32 %0, %1, %2"
                   :
                   : "v"(loff), "v"(goff), "s"(xbase)
                   : "memory");
    }
    asm volatile("s_wait_asynccnt 0x0" ::: "memory");
  }
  __syncthreads();

  // ---- stage 2: convert once into fragment-ordered bf16 ----
  // element (k,m): kk=k>>5, r=k&31, hk=(r>>4)&1, halfk=(r>>3)&1, i=r&7
  // dest = (((kk*2 + hk)*32 + m)*2 + halfk)*8 + i
  for (int idx = tid; idx < 4096; idx += 256) {
    int k = idx >> 5, m = idx & 31;
    int kk = k >> 5, r = k & 31;
    int halfk = (r >> 3) & 1, hk = (r >> 4) & 1, i = r & 7;
    int dest = (((kk * 2 + hk) * 32 + m) * 2 + halfk) * 8 + i;
    sAb[dest] = to_bf16(sAf[idx]);
  }
  __syncthreads();

  const int lane = tid & 31, wave = tid >> 5;
  const int hi = lane >> 4, ln = lane & 15;
  const int e0 = half * 128 + wave * 16;    // output column base (0..255)

  v8f acc0 = {}, acc1 = {};
#pragma unroll
  for (int kk = 0; kk < 4; ++kk) {
    const int k0 = kk * 32;
    // B frag, k order chosen here (A matches): column n=ln, K = 16*hi + i -> one contiguous 32B load
    const v16bf bfrag = *(const v16bf*)(wb + (size_t)(e0 + ln) * 128 + k0 + hi * 16);
    // A frags: contiguous 16 bf16 per lane (fragment-ordered LDS)
    const v16bf a0 = *(const v16bf*)(&sAb[((kk * 2 + hi) * 32 + ln) * 16]);
    const v16bf a1 = *(const v16bf*)(&sAb[((kk * 2 + hi) * 32 + 16 + ln) * 16]);
    acc0 = __builtin_amdgcn_wmma_f32_16x16x32_bf16(false, a0, false, bfrag,
                                                   (short)0, acc0, false, false);
    acc1 = __builtin_amdgcn_wmma_f32_16x16x32_bf16(false, a1, false, bfrag,
                                                   (short)0, acc1, false, false);
  }

  if (half == 0) {
    // D frag element j: m = hi*8+j (+16 for tile 1), n = ln -> 8 contiguous floats
    float* dst = xin_ch + ((size_t)b * D_ + (e0 + ln)) * L_ + l0;
#pragma unroll
    for (int j = 0; j < 8; ++j) dst[hi * 8 + j]      = acc0[j];
#pragma unroll
    for (int j = 0; j < 8; ++j) dst[16 + hi * 8 + j] = acc1[j];
  } else {
    const int ez = e0 - 128 + ln;
#pragma unroll
    for (int j = 0; j < 8; ++j) {
      z_tok[((size_t)(g0 + hi * 8 + j)) * D_ + ez]      = fast_silu(acc0[j]);
      z_tok[((size_t)(g0 + 16 + hi * 8 + j)) * D_ + ez] = fast_silu(acc1[j]);
    }
  }
}

// =====================================================================
// K2: depthwise causal conv (width 4, left pad 3) + bias + SiLU
// =====================================================================
__global__ __launch_bounds__(256) void k_conv(const float* __restrict__ xin_ch,
                                              const float* __restrict__ cw,
                                              const float* __restrict__ cb,
                                              float* __restrict__ xc_ch) {
  const int bd = blockIdx.x;
  const int b = bd >> 7, d = bd & 127;
  const float w0 = cw[d * 4 + 0], w1 = cw[d * 4 + 1];
  const float w2 = cw[d * 4 + 2], w3 = cw[d * 4 + 3];
  const float bias = cb[d];
  const float* src = xin_ch + ((size_t)b * D_ + d) * L_;
  float* dst       = xc_ch  + ((size_t)b * D_ + d) * L_;
  for (int l = threadIdx.x; l < L_; l += 256) {
    float x3 = src[l];
    float x2 = (l >= 1) ? src[l - 1] : 0.0f;
    float x1 = (l >= 2) ? src[l - 2] : 0.0f;
    float x0 = (l >= 3) ? src[l - 3] : 0.0f;
    float c = w0 * x0 + w1 * x1 + w2 * x2 + w3 * x3 + bias;
    dst[l] = fast_silu(c);
  }
}

// =====================================================================
// K3: per token: x_dbl = x_proj(xc); dt = softplus(dt_proj(x_dbl[:8])+b)
//     also emit token-major copy of xc. One thread per token.
// =====================================================================
__global__ __launch_bounds__(256) void k_proj(const float* __restrict__ xc_ch,
                                              const float* __restrict__ xpw,
                                              const float* __restrict__ dtw,
                                              const float* __restrict__ dtb,
                                              float* __restrict__ xc_tok,
                                              float* __restrict__ dt_tok,
                                              float* __restrict__ Bc,
                                              float* __restrict__ Cc) {
  __shared__ float s_xpw[12 * 128];
  __shared__ float s_dtw[128 * DTRANK];
  __shared__ float s_dtb[128];
  const int tid = threadIdx.x;
  for (int i = tid; i < 12 * 128; i += 256) s_xpw[i] = xpw[i];
  for (int i = tid; i < 128 * DTRANK; i += 256) s_dtw[i] = dtw[i];
  if (tid < 128) s_dtb[tid] = dtb[tid];
  __syncthreads();

  const int g = blockIdx.x * 256 + tid;   // token id
  const int b = g / L_, l = g % L_;
  float acc[12];
#pragma unroll
  for (int e = 0; e < 12; ++e) acc[e] = 0.0f;

  const float* col = xc_ch + (size_t)b * D_ * L_ + l;
  for (int d = 0; d < 128; ++d) {
    float xv = col[(size_t)d * L_];
    xc_tok[(size_t)g * D_ + d] = xv;
#pragma unroll
    for (int e = 0; e < 12; ++e) acc[e] += xv * s_xpw[e * 128 + d];
  }
  Bc[(size_t)g * 2 + 0] = acc[DTRANK + 0];
  Bc[(size_t)g * 2 + 1] = acc[DTRANK + 1];
  Cc[(size_t)g * 2 + 0] = acc[DTRANK + 2];
  Cc[(size_t)g * 2 + 1] = acc[DTRANK + 3];

  for (int d = 0; d < 128; ++d) {
    float s = s_dtb[d];
#pragma unroll
    for (int r = 0; r < DTRANK; ++r) s += acc[r] * s_dtw[d * DTRANK + r];
    float sp = (s > 20.0f) ? s : __logf(1.0f + __expf(s));
    dt_tok[(size_t)g * D_ + d] = sp;
  }
}

// =====================================================================
// K4a: per-chunk scan aggregates (Aprod, Bsum) per (d,s)
// =====================================================================
__global__ __launch_bounds__(32) void k_scan_agg(const float* __restrict__ dt_tok,
                                                 const float* __restrict__ xc_tok,
                                                 const float* __restrict__ Bc,
                                                 const float* __restrict__ Alog,
                                                 float* __restrict__ aggA,
                                                 float* __restrict__ aggB) {
  const int blk = blockIdx.x;
  const int b = blk / NCHUNK, ch = blk - b * NCHUNK;
  const int lane = threadIdx.x, d0 = lane * 4;
  float Av[4][2], hA[4][2], hB[4][2];
#pragma unroll
  for (int i = 0; i < 4; ++i) {
    Av[i][0] = -__expf(Alog[(d0 + i) * 2 + 0]);
    Av[i][1] = -__expf(Alog[(d0 + i) * 2 + 1]);
    hA[i][0] = hA[i][1] = 1.0f;
    hB[i][0] = hB[i][1] = 0.0f;
  }
  const size_t tokbase = (size_t)b * L_ + ch * CL;
  for (int t = 0; t < CL; ++t) {
    size_t tok = tokbase + t;
    float4 dt4 = *(const float4*)(dt_tok + tok * D_ + d0);
    float4 xc4 = *(const float4*)(xc_tok + tok * D_ + d0);
    float b0 = Bc[tok * 2 + 0], b1 = Bc[tok * 2 + 1];
    float dts[4] = {dt4.x, dt4.y, dt4.z, dt4.w};
    float xcs[4] = {xc4.x, xc4.y, xc4.z, xc4.w};
#pragma unroll
    for (int i = 0; i < 4; ++i) {
      float a0 = __expf(dts[i] * Av[i][0]);
      float a1 = __expf(dts[i] * Av[i][1]);
      hB[i][0] = a0 * hB[i][0] + dts[i] * b0 * xcs[i];
      hB[i][1] = a1 * hB[i][1] + dts[i] * b1 * xcs[i];
      hA[i][0] *= a0;
      hA[i][1] *= a1;
    }
  }
  const size_t base = (size_t)blk * 256 + lane * 8;
#pragma unroll
  for (int i = 0; i < 4; ++i) {
    aggA[base + i * 2 + 0] = hA[i][0];
    aggA[base + i * 2 + 1] = hA[i][1];
    aggB[base + i * 2 + 0] = hB[i][0];
    aggB[base + i * 2 + 1] = hB[i][1];
  }
}

// K4b: sequential prefix over the 64 chunks (exclusive init state per chunk)
__global__ __launch_bounds__(256) void k_scan_pref(const float* __restrict__ aggA,
                                                   const float* __restrict__ aggB,
                                                   float* __restrict__ hinit) {
  const int b = blockIdx.x, t = threadIdx.x;   // t = (d,s) index 0..255
  float h = 0.0f;
  for (int c = 0; c < NCHUNK; ++c) {
    size_t i = ((size_t)(b * NCHUNK + c)) * 256 + t;
    hinit[i] = h;
    h = aggA[i] * h + aggB[i];
  }
}

// K4c: re-scan with init state; fuse y = C.h + D*x, y *= silu(z); emit bf16 y
__global__ __launch_bounds__(32) void k_scan_apply(const float* __restrict__ dt_tok,
                                                   const float* __restrict__ xc_tok,
                                                   const float* __restrict__ Bc,
                                                   const float* __restrict__ Cc,
                                                   const float* __restrict__ z_tok,
                                                   const float* __restrict__ Alog,
                                                   const float* __restrict__ Dp,
                                                   const float* __restrict__ hinit,
                                                   bf16* __restrict__ y_bf) {
  const int blk = blockIdx.x;
  const int b = blk / NCHUNK, ch = blk - b * NCHUNK;
  const int lane = threadIdx.x, d0 = lane * 4;
  float Av[4][2], Dv[4], h[4][2];
#pragma unroll
  for (int i = 0; i < 4; ++i) {
    Av[i][0] = -__expf(Alog[(d0 + i) * 2 + 0]);
    Av[i][1] = -__expf(Alog[(d0 + i) * 2 + 1]);
    Dv[i] = Dp[d0 + i];
    h[i][0] = hinit[(size_t)blk * 256 + lane * 8 + i * 2 + 0];
    h[i][1] = hinit[(size_t)blk * 256 + lane * 8 + i * 2 + 1];
  }
  const size_t tokbase = (size_t)b * L_ + ch * CL;
  for (int t = 0; t < CL; ++t) {
    size_t tok = tokbase + t;
    float4 dt4 = *(const float4*)(dt_tok + tok * D_ + d0);
    float4 xc4 = *(const float4*)(xc_tok + tok * D_ + d0);
    float4 zs4 = *(const float4*)(z_tok + tok * D_ + d0);
    float b0 = Bc[tok * 2 + 0], b1 = Bc[tok * 2 + 1];
    float c0 = Cc[tok * 2 + 0], c1 = Cc[tok * 2 + 1];
    float dts[4] = {dt4.x, dt4.y, dt4.z, dt4.w};
    float xcs[4] = {xc4.x, xc4.y, xc4.z, xc4.w};
    float zss[4] = {zs4.x, zs4.y, zs4.z, zs4.w};
    v4bf yv;
#pragma unroll
    for (int i = 0; i < 4; ++i) {
      float a0 = __expf(dts[i] * Av[i][0]);
      float a1 = __expf(dts[i] * Av[i][1]);
      h[i][0] = a0 * h[i][0] + dts[i] * b0 * xcs[i];
      h[i][1] = a1 * h[i][1] + dts[i] * b1 * xcs[i];
      float y = (h[i][0] * c0 + h[i][1] * c1 + Dv[i] * xcs[i]) * zss[i];
      yv[i] = to_bf16(y);
    }
    *(v4bf*)(y_bf + tok * D_ + d0) = yv;     // 8B aligned store
  }
}

// =====================================================================
// K5: out = y @ out_proj^T (WMMA bf16, all-bf16 operand loads) + LayerNorm
// =====================================================================
__global__ __launch_bounds__(256) void k_out_ln(const bf16* __restrict__ y_bf,
                                                const bf16* __restrict__ wb,
                                                const float* __restrict__ gamma,
                                                const float* __restrict__ beta,
                                                float* __restrict__ out) {
  __shared__ float sOut[16 * 128];
  __shared__ float sMu[16], sRs[16];
  const int g0 = blockIdx.x * 16;
  const int b = g0 / L_, l0 = g0 % L_;
  const int tid = threadIdx.x, lane = tid & 31, wave = tid >> 5;
  const int hi = lane >> 4, ln = lane & 15;
  const int n0 = wave * 16;

  v8f acc = {};
#pragma unroll
  for (int kk = 0; kk < 4; ++kk) {
    const int k0 = kk * 32;
    const bf16* arow = y_bf + (size_t)(g0 + ln) * D_ + k0;     // row m=ln
    __builtin_prefetch(arow + 64, 0, 1);
    v8bf alo = *(const v8bf*)(arow + hi * 16);                 // K = hi*16 + 0..7
    v8bf ahi = *(const v8bf*)(arow + hi * 16 + 8);             // K = hi*16+8 + 0..7
    v16bf afrag = __builtin_shufflevector(alo, ahi,
        0, 1, 2, 3, 4, 5, 6, 7, 8, 9, 10, 11, 12, 13, 14, 15);
    const v16bf bfrag = *(const v16bf*)(wb + (size_t)(n0 + ln) * 128 + k0 + hi * 16);
    acc = __builtin_amdgcn_wmma_f32_16x16x32_bf16(false, afrag, false, bfrag,
                                                  (short)0, acc, false, false);
  }
#pragma unroll
  for (int j = 0; j < 8; ++j) sOut[(hi * 8 + j) * 128 + n0 + ln] = acc[j];
  __syncthreads();

  if (tid < 16) {
    float mu = 0.0f;
    for (int n = 0; n < 128; ++n) mu += sOut[tid * 128 + n];
    mu *= (1.0f / 128.0f);
    float var = 0.0f;
    for (int n = 0; n < 128; ++n) {
      float d = sOut[tid * 128 + n] - mu;
      var += d * d;
    }
    var *= (1.0f / 128.0f);
    sMu[tid] = mu;
    sRs[tid] = rsqrtf(var + 1e-5f);
  }
  __syncthreads();

  for (int idx = tid; idx < 2048; idx += 256) {
    int m = idx & 15, n = idx >> 4;
    float v = (sOut[m * 128 + n] - sMu[m]) * sRs[m] * gamma[n] + beta[n];
    out[((size_t)b * OUTC + n) * L_ + (l0 + m)] = v;   // (B, OUT_C, H, W)
  }
}

// =====================================================================
extern "C" void kernel_launch(void* const* d_in, const int* in_sizes, int n_in,
                              void* d_out, int out_size, void* d_ws, size_t ws_size,
                              hipStream_t stream) {
  const float* x      = (const float*)d_in[0];
  const float* w_in   = (const float*)d_in[1];
  const float* conv_w = (const float*)d_in[2];
  const float* conv_b = (const float*)d_in[3];
  const float* xpw    = (const float*)d_in[4];
  const float* dtw    = (const float*)d_in[5];
  const float* dtb    = (const float*)d_in[6];
  const float* Alog   = (const float*)d_in[7];
  const float* Dp     = (const float*)d_in[8];
  const float* ow     = (const float*)d_in[9];
  const float* gamma  = (const float*)d_in[10];
  const float* beta   = (const float*)d_in[11];
  float* out = (float*)d_out;
  float* ws  = (float*)d_ws;

  const size_t F = (size_t)B_ * D_ * L_;     // 18,874,368 floats
  float* xin_ch = ws;                        // (B,D,L); reused as xc_tok after K2
  float* z_tok  = ws + F;                    // (B*L,D) silu(z)
  float* xc_ch  = ws + 2 * F;                // (B,D,L); region reused for bf16 y after K3
  float* dt_tok = ws + 3 * F;                // (B*L,D)
  float* Bc     = ws + 4 * F;                // (B*L,2)
  float* Cc     = Bc + (size_t)NTOK * 2;     // (B*L,2)
  float* aggA   = Cc + (size_t)NTOK * 2;     // (B*NCHUNK,256)
  float* aggB   = aggA + (size_t)B_ * NCHUNK * 256;
  float* hinit  = aggB + (size_t)B_ * NCHUNK * 256;
  bf16* wb_in   = (bf16*)(hinit + (size_t)B_ * NCHUNK * 256);  // 256x128 bf16
  bf16* wb_out  = wb_in + 256 * 128;                           // 128x128 bf16
  float* xc_tok = xin_ch;                    // alias: x_in dead after K2
  bf16*  y_bf   = (bf16*)xc_ch;              // alias: xc_ch dead after K3

  k_prep      <<<128,                256, 0, stream>>>(w_in, ow, wb_in, wb_out);
  k_in_proj   <<<dim3(NTOK / 32, 2), 256, 0, stream>>>(x, wb_in, xin_ch, z_tok);
  k_conv      <<<B_ * D_,            256, 0, stream>>>(xin_ch, conv_w, conv_b, xc_ch);
  k_proj      <<<NTOK / 256,         256, 0, stream>>>(xc_ch, xpw, dtw, dtb,
                                                       xc_tok, dt_tok, Bc, Cc);
  k_scan_agg  <<<B_ * NCHUNK,        32,  0, stream>>>(dt_tok, xc_tok, Bc, Alog, aggA, aggB);
  k_scan_pref <<<B_,                 256, 0, stream>>>(aggA, aggB, hinit);
  k_scan_apply<<<B_ * NCHUNK,        32,  0, stream>>>(dt_tok, xc_tok, Bc, Cc, z_tok,
                                                       Alog, Dp, hinit, y_bf);
  k_out_ln    <<<NTOK / 16,          256, 0, stream>>>(y_bf, wb_out, gamma, beta, out);
}
